// GraphNeuralNetwork_51316269253151
// MI455X (gfx1250) — matrix operand order repaired and checked by reading, not compiled, not run on hardware
//
#include <hip/hip_runtime.h>

// ---------------------------------------------------------------------------
// Types for WMMA (CDNA5 / gfx1250, wave32)
// ---------------------------------------------------------------------------
typedef _Float16 v16h __attribute__((ext_vector_type(16)));
typedef _Float16 v8h  __attribute__((ext_vector_type(8)));
typedef float    v8f  __attribute__((ext_vector_type(8)));

#define LRELU_SLOPE 0.2f

// ---------------------------------------------------------------------------
// gfx1250 async global->LDS copy (16B per lane), ASYNCcnt-tracked
// ---------------------------------------------------------------------------
__device__ __forceinline__ void async_cp16(unsigned lds_off, const _Float16* g) {
    asm volatile("global_load_async_to_lds_b128 %0, %1, off"
                 :: "v"(lds_off), "v"(g) : "memory");
}
__device__ __forceinline__ void wait_async0() {
#if defined(__has_builtin)
#if __has_builtin(__builtin_amdgcn_s_wait_asynccnt)
    __builtin_amdgcn_s_wait_asynccnt(0);
#else
    asm volatile("s_wait_asynccnt 0x0" ::: "memory");
#endif
#else
    asm volatile("s_wait_asynccnt 0x0" ::: "memory");
#endif
}

// ---------------------------------------------------------------------------
// Utility kernels
// ---------------------------------------------------------------------------
__global__ void fill_f32(float* __restrict__ p, float v, long n) {
    long i = (long)blockIdx.x * blockDim.x + threadIdx.x;
    if (i < n) p[i] = v;
}

__global__ void fill_u32(unsigned* __restrict__ p, unsigned v, long n) {
    long i = (long)blockIdx.x * blockDim.x + threadIdx.x;
    if (i < n) p[i] = v;
}

__global__ void relu_k(float* __restrict__ p, long n) {
    long i = (long)blockIdx.x * blockDim.x + threadIdx.x;
    if (i < n) p[i] = fmaxf(p[i], 0.0f);
}

__global__ void init_bias(float* __restrict__ out, const float* __restrict__ b,
                          long n, int H) {
    long i = (long)blockIdx.x * blockDim.x + threadIdx.x;
    if (i < n) out[i] = b[(int)(i % H)];
}

// f32 -> f16 with zero padding (rows and columns)
__global__ void cvt_pad2d(const float* __restrict__ in, _Float16* __restrict__ out,
                          int rIn, int cIn, int rOut, int cOut) {
    long i = (long)blockIdx.x * blockDim.x + threadIdx.x;
    long total = (long)rOut * cOut;
    if (i >= total) return;
    int r = (int)(i / cOut), c = (int)(i % cOut);
    float v = (r < rIn && c < cIn) ? in[(long)r * cIn + c] : 0.0f;
    out[i] = (_Float16)v;
}

// ---------------------------------------------------------------------------
// WMMA GEMM: C[M,N] = A[M,K](f16) @ B[K,N](f16), f32 accumulate.
// K multiple of 32, N multiple of 128. Block = 256 threads = 8 waves (wave32).
// Workgroup tile 128x128; each wave computes 64x32 via 4x2 WMMA 16x16 tiles.
// A/B staged through LDS with gfx1250 GLOBAL_LOAD_ASYNC_TO_LDS_B128.
// ---------------------------------------------------------------------------
#define SA_STRIDE 40    // 32 halves + 8 pad  (row base 80B, 16B aligned)
#define SB_STRIDE 136   // 128 halves + 8 pad (row base 272B, 16B aligned)

__global__ __launch_bounds__(256)
void gemm_f16_wmma(const _Float16* __restrict__ A, const _Float16* __restrict__ B,
                   float* __restrict__ C, int M, int N, int K) {
    __shared__ __align__(16) _Float16 sA[128 * SA_STRIDE];  // 10.0 KB
    __shared__ __align__(16) _Float16 sB[32 * SB_STRIDE];   // 8.5 KB

    const int tid    = threadIdx.x;
    const int lane   = tid & 31;
    const int wave   = tid >> 5;        // 0..7
    const int wm     = wave >> 2;       // 0..1 -> row halves of 64
    const int wn     = wave & 3;        // 0..3 -> col quarters of 32
    const int lanelo = lane & 15;
    const int hi     = lane >> 4;       // 0/1 half-wave
    const int bM     = blockIdx.x * 128;
    const int bN     = blockIdx.y * 128;
    const bool full  = (bM + 128 <= M); // uniform: all rows in-bounds?

    v8f acc[4][2] = {};

    // ISA 7.12.2: 16-bit A 16x32 layout. Element e of the v16h fragment maps to
    // K = base(j) + 8*hi + p where j=e>>1, p=e&1, base = {0,2,4,6,16,18,20,22}.
    int kmap[16];
#pragma unroll
    for (int e = 0; e < 16; ++e) {
        int j = e >> 1, p1 = e & 1;
        kmap[e] = ((j < 4) ? (2 * j) : (16 + 2 * (j - 4))) + 8 * hi + p1;
    }

    const int nk = K / 32;
    for (int kt = 0; kt < nk; ++kt) {
        if (full) {
            // --- async stage A tile: 128x32 halves = 512 chunks of 16B -------
#pragma unroll
            for (int it = 0; it < 2; ++it) {
                int ch = tid + it * 256;
                int r = ch >> 2, c8 = (ch & 3) * 8;
                unsigned lo = (unsigned)(size_t)&sA[r * SA_STRIDE + c8];
                async_cp16(lo, A + (long)(bM + r) * K + (long)kt * 32 + c8);
            }
            // --- async stage B tile: 32x128 halves = 512 chunks of 16B ------
#pragma unroll
            for (int it = 0; it < 2; ++it) {
                int ch = tid + it * 256;
                int r = ch >> 4, c8 = (ch & 15) * 8;
                unsigned lo = (unsigned)(size_t)&sB[r * SB_STRIDE + c8];
                async_cp16(lo, B + (long)(kt * 32 + r) * N + bN + c8);
            }
            wait_async0();
        } else {
            // --- tail row-block: guarded loads with zero fill ---------------
#pragma unroll
            for (int it = 0; it < 2; ++it) {
                int ch = tid + it * 256;
                int r = ch >> 2, c8 = (ch & 3) * 8;
                uint4 v = make_uint4(0u, 0u, 0u, 0u);
                int gr = bM + r;
                if (gr < M)
                    v = *(const uint4*)(A + (long)gr * K + (long)kt * 32 + c8);
                *(uint4*)(&sA[r * SA_STRIDE + c8]) = v;
            }
#pragma unroll
            for (int it = 0; it < 2; ++it) {
                int ch = tid + it * 256;
                int r = ch >> 4, c8 = (ch & 15) * 8;
                uint4 v = *(const uint4*)(B + (long)(kt * 32 + r) * N + bN + c8);
                *(uint4*)(&sB[r * SB_STRIDE + c8]) = v;
            }
        }
        __syncthreads();

        // Prefetch next A k-tile into cache (gfx1250 global_prefetch_b8)
        if (full && kt + 1 < nk)
            __builtin_prefetch(A + (long)(bM + (tid >> 1)) * K + (long)(kt + 1) * 32, 0, 1);

        v16h af[4], bf[2];
#pragma unroll
        for (int i = 0; i < 4; ++i) {
            int r = wm * 64 + i * 16 + lanelo;
#pragma unroll
            for (int e = 0; e < 16; ++e) af[i][e] = sA[r * SA_STRIDE + kmap[e]];
        }
#pragma unroll
        for (int j = 0; j < 2; ++j) {
            int cc = wn * 32 + j * 16;
#pragma unroll
            for (int e = 0; e < 16; ++e) bf[j][e] = sB[kmap[e] * SB_STRIDE + cc + lanelo];
        }

#pragma unroll
        for (int i = 0; i < 4; ++i)
#pragma unroll
            for (int j = 0; j < 2; ++j)
                acc[i][j] = __builtin_amdgcn_wmma_f32_16x16x32_f16(
                    false, af[i], false, bf[j], (short)0, acc[i][j], false, false);
        __syncthreads();
    }

    // C/D layout (ISA 7.12.2): VGPR r, lanes 0-15: M=r; lanes 16-31: M=r+8
#pragma unroll
    for (int i = 0; i < 4; ++i) {
#pragma unroll
        for (int j = 0; j < 2; ++j) {
            int col = bN + wn * 32 + j * 16 + lanelo;
#pragma unroll
            for (int r = 0; r < 8; ++r) {
                int row = bM + wm * 64 + i * 16 + hi * 8 + r;
                if (row < M) C[(long)row * N + col] = acc[i][j][r];
            }
        }
    }
}

// ---------------------------------------------------------------------------
// GCN pieces
// ---------------------------------------------------------------------------
__global__ void deg_accum(const int* __restrict__ dst, float* __restrict__ deg, int E) {
    int i = blockIdx.x * blockDim.x + threadIdx.x;
    if (i < E) atomicAdd(&deg[dst[i]], 1.0f);
}

__global__ void rsqrt_k(const float* __restrict__ deg, float* __restrict__ dinv, int n) {
    int i = blockIdx.x * blockDim.x + threadIdx.x;
    if (i < n) dinv[i] = rsqrtf(deg[i]);
}

// out = bias + h * dinv^2   (self-loop contribution folded in)
__global__ void gcn_init(const float* __restrict__ h, const float* __restrict__ dinv,
                         const float* __restrict__ b, float* __restrict__ out,
                         int N, int H) {
    long i = (long)blockIdx.x * blockDim.x + threadIdx.x;
    if (i >= (long)N * H) return;
    int n = (int)(i / H), c = (int)(i % H);
    float di = dinv[n];
    out[i] = b[c] + h[i] * di * di;
}

__global__ void gcn_scatter(const float* __restrict__ h, const int* __restrict__ src,
                            const int* __restrict__ dst, const float* __restrict__ dinv,
                            float* __restrict__ out, int H) {
    int e = blockIdx.x;
    int s = src[e], d = dst[e];
    float norm = dinv[s] * dinv[d];
    for (int c = threadIdx.x; c < H; c += blockDim.x)
        atomicAdd(&out[(long)d * H + c], h[(long)s * H + c] * norm);
}

// ---------------------------------------------------------------------------
// GAT pieces
// ---------------------------------------------------------------------------
__device__ __forceinline__ unsigned fenc(float f) {
    unsigned u = __float_as_uint(f);
    return (u & 0x80000000u) ? ~u : (u | 0x80000000u);
}
__device__ __forceinline__ float fdec(unsigned u) {
    unsigned b = (u & 0x80000000u) ? (u & 0x7FFFFFFFu) : ~u;
    return __uint_as_float(b);
}
__device__ __forceinline__ void edge_sd(const int* src, const int* dst, int E,
                                        int idx, int& s, int& d) {
    if (idx < E) { s = src[idx]; d = dst[idx]; }
    else         { s = d = idx - E; }          // self-loops appended
}

__global__ void att_dots(const float* __restrict__ h, const float* __restrict__ as_,
                         const float* __restrict__ ad_, float* __restrict__ al_s,
                         float* __restrict__ al_d, int N, int NH, int HD, int H) {
    int i = blockIdx.x * blockDim.x + threadIdx.x;
    if (i >= N * NH) return;
    int n = i / NH, hh = i % NH;
    const float* hp = h + (long)n * H + hh * HD;
    const float* ap = as_ + hh * HD;
    const float* bp = ad_ + hh * HD;
    float s = 0.f, t = 0.f;
    for (int k = 0; k < HD; ++k) { float v = hp[k]; s += v * ap[k]; t += v * bp[k]; }
    al_s[i] = s; al_d[i] = t;
}

__global__ void edge_logits(const int* __restrict__ src, const int* __restrict__ dst,
                            int E, int EE, const float* __restrict__ al_s,
                            const float* __restrict__ al_d, float* __restrict__ ebuf,
                            unsigned* __restrict__ mx, int NH) {
    long i = (long)blockIdx.x * blockDim.x + threadIdx.x;
    if (i >= (long)EE * NH) return;
    int idx = (int)(i / NH), hh = (int)(i % NH);
    int s, d; edge_sd(src, dst, E, idx, s, d);
    float ev = al_s[s * NH + hh] + al_d[d * NH + hh];
    ev = (ev > 0.f) ? ev : LRELU_SLOPE * ev;
    ebuf[i] = ev;
    atomicMax(&mx[d * NH + hh], fenc(ev));
}

__global__ void edge_expsum(const int* __restrict__ src, const int* __restrict__ dst,
                            int E, int EE, float* __restrict__ ebuf,
                            const unsigned* __restrict__ mx, float* __restrict__ ssum,
                            int NH) {
    long i = (long)blockIdx.x * blockDim.x + threadIdx.x;
    if (i >= (long)EE * NH) return;
    int idx = (int)(i / NH), hh = (int)(i % NH);
    int s, d; edge_sd(src, dst, E, idx, s, d);
    (void)s;
    float m  = fdec(mx[d * NH + hh]);
    float ex = expf(ebuf[i] - m);
    ebuf[i] = ex;
    atomicAdd(&ssum[d * NH + hh], ex);
}

__global__ void gat_scatter(const float* __restrict__ h, const int* __restrict__ src,
                            const int* __restrict__ dst, int E,
                            const float* __restrict__ ebuf, const float* __restrict__ ssum,
                            float* __restrict__ out, int H, int NH) {
    int idx = blockIdx.x;
    int s, d; edge_sd(src, dst, E, idx, s, d);
    for (int c = threadIdx.x; c < H; c += blockDim.x) {
        int hh = c >> 6;                  // HD = 64
        float alpha = ebuf[(long)idx * NH + hh] / ssum[d * NH + hh];
        atomicAdd(&out[(long)d * H + c], h[(long)s * H + c] * alpha);
    }
}

// ---------------------------------------------------------------------------
// Pooling + classifier
// ---------------------------------------------------------------------------
__global__ void pool_cnt(const int* __restrict__ batch, float* __restrict__ cnt, int N) {
    int i = blockIdx.x * blockDim.x + threadIdx.x;
    if (i < N) atomicAdd(&cnt[batch[i]], 1.0f);
}

__global__ void pool_sum(const float* __restrict__ h, const int* __restrict__ batch,
                         float* __restrict__ g, int N, int H) {
    long i = (long)blockIdx.x * blockDim.x + threadIdx.x;
    if (i >= (long)N * H) return;
    int n = (int)(i / H), c = (int)(i % H);
    atomicAdd(&g[(long)batch[n] * H + c], h[i]);
}

__global__ void pool_div(float* __restrict__ g, const float* __restrict__ cnt,
                         int NG, int H) {
    int i = blockIdx.x * blockDim.x + threadIdx.x;
    if (i >= NG * H) return;
    g[i] /= fmaxf(cnt[i / H], 1.0f);
}

__global__ void small_gemm(const float* __restrict__ A, const float* __restrict__ W,
                           const float* __restrict__ b, float* __restrict__ out,
                           int M, int K, int Nc, int relu) {
    int i = blockIdx.x * blockDim.x + threadIdx.x;
    if (i >= M * Nc) return;
    int m = i / Nc, n = i % Nc;
    float acc = b[n];
    for (int k = 0; k < K; ++k) acc += A[(long)m * K + k] * W[(long)k * Nc + n];
    out[i] = relu ? fmaxf(acc, 0.f) : acc;
}

// ---------------------------------------------------------------------------
// Launcher
// ---------------------------------------------------------------------------
extern "C" void kernel_launch(void* const* d_in, const int* in_sizes, int n_in,
                              void* d_out, int out_size, void* d_ws, size_t ws_size,
                              hipStream_t stream) {
    (void)n_in; (void)out_size; (void)ws_size;

    const float* x     = (const float*)d_in[0];
    const int*   ei    = (const int*)d_in[1];
    const int*   batch = (const int*)d_in[2];
    const float* w1 = (const float*)d_in[3];  const float* b1 = (const float*)d_in[4];
    const float* w2 = (const float*)d_in[5];  const float* b2 = (const float*)d_in[6];
    const float* w3 = (const float*)d_in[7];  const float* b3 = (const float*)d_in[8];
    const float* wg1 = (const float*)d_in[9];  const float* as1 = (const float*)d_in[10];
    const float* ad1 = (const float*)d_in[11]; const float* bg1 = (const float*)d_in[12];
    const float* wg2 = (const float*)d_in[13]; const float* as2 = (const float*)d_in[14];
    const float* ad2 = (const float*)d_in[15]; const float* bg2 = (const float*)d_in[16];
    const float* wc1 = (const float*)d_in[17]; const float* bc1 = (const float*)d_in[18];
    const float* wc2 = (const float*)d_in[19]; const float* bc2 = (const float*)d_in[20];

    const int N  = in_sizes[0] / 3;
    const int E  = in_sizes[1] / 2;
    const int H  = 768, NH = 12, HD = 64, NG = 64;
    const int Hc = in_sizes[18];   // 384
    const int NC = in_sizes[20];   // 10
    const int EE = E + N;
    const int* srcp = ei;
    const int* dstp = ei + E;

    // workspace bump allocator (256B aligned)
    char* p = (char*)d_ws;
    auto bump = [&](size_t bytes) -> void* {
        char* r = p; p += (bytes + 255) & ~(size_t)255; return (void*)r;
    };
    _Float16* actX = (_Float16*)bump((size_t)N * H * 2);
    _Float16* wbuf = (_Float16*)bump((size_t)H * H * 2);
    float*    bufA = (float*)bump((size_t)N * H * 4);
    float*    bufB = (float*)bump((size_t)N * H * 4);
    float*    deg  = (float*)bump((size_t)N * 4);
    float*    dinv = (float*)bump((size_t)N * 4);
    float*    als  = (float*)bump((size_t)N * NH * 4);
    float*    ald  = (float*)bump((size_t)N * NH * 4);
    unsigned* mx   = (unsigned*)bump((size_t)N * NH * 4);
    float*    ssum = (float*)bump((size_t)N * NH * 4);
    float*    ebuf = (float*)bump((size_t)EE * NH * 4);
    float*    g    = (float*)bump((size_t)NG * H * 4);
    float*    cnt  = (float*)bump((size_t)NG * 4);
    float*    z    = (float*)bump((size_t)NG * Hc * 4);

    auto blocks = [](long n) { return dim3((unsigned)((n + 255) / 256)); };

    // Degrees (shared across GCN layers); deg starts at 1.0 for self-loop.
    fill_f32<<<blocks(N), 256, 0, stream>>>(deg, 1.0f, N);
    deg_accum<<<blocks(E), 256, 0, stream>>>(dstp, deg, E);
    rsqrt_k<<<blocks(N), 256, 0, stream>>>(deg, dinv, N);

    // GEMM via WMMA: out = Xin @ W   (f16 inputs, f32 out); colsW % 128 == 0
    auto gemm = [&](const float* Xin, int colsX, const float* W, float* outC, int colsW) {
        int Kp = ((colsX + 31) / 32) * 32;
        cvt_pad2d<<<blocks((long)N * Kp), 256, 0, stream>>>(Xin, actX, N, colsX, N, Kp);
        cvt_pad2d<<<blocks((long)Kp * colsW), 256, 0, stream>>>(W, wbuf, colsX, colsW, Kp, colsW);
        gemm_f16_wmma<<<dim3((N + 127) / 128, colsW / 128), 256, 0, stream>>>(
            actX, wbuf, outC, N, colsW, Kp);
    };

    auto gcn = [&](const float* xin, int cin, const float* W, const float* b, bool relu) {
        gemm(xin, cin, W, bufA, H);
        gcn_init<<<blocks((long)N * H), 256, 0, stream>>>(bufA, dinv, b, bufB, N, H);
        gcn_scatter<<<dim3(E), 256, 0, stream>>>(bufA, srcp, dstp, dinv, bufB, H);
        if (relu) relu_k<<<blocks((long)N * H), 256, 0, stream>>>(bufB, (long)N * H);
    };

    auto gat = [&](const float* xin, const float* W, const float* as_, const float* ad_,
                   const float* b, bool relu) {
        gemm(xin, H, W, bufA, H);
        att_dots<<<blocks((long)N * NH), 256, 0, stream>>>(bufA, as_, ad_, als, ald, N, NH, HD, H);
        fill_u32<<<blocks((long)N * NH), 256, 0, stream>>>(mx, 0x007FFFFFu, (long)N * NH); // enc(-inf)
        fill_f32<<<blocks((long)N * NH), 256, 0, stream>>>(ssum, 0.0f, (long)N * NH);
        edge_logits<<<blocks((long)EE * NH), 256, 0, stream>>>(srcp, dstp, E, EE, als, ald, ebuf, mx, NH);
        edge_expsum<<<blocks((long)EE * NH), 256, 0, stream>>>(srcp, dstp, E, EE, ebuf, mx, ssum, NH);
        init_bias<<<blocks((long)N * H), 256, 0, stream>>>(bufB, b, (long)N * H, H);
        gat_scatter<<<dim3(EE), 256, 0, stream>>>(bufA, srcp, dstp, E, ebuf, ssum, bufB, H, NH);
        if (relu) relu_k<<<blocks((long)N * H), 256, 0, stream>>>(bufB, (long)N * H);
    };

    // --- network ---
    gcn(x,    3, w1, b1, true);
    gcn(bufB, H, w2, b2, true);
    gcn(bufB, H, w3, b3, true);
    gat(bufB, wg1, as1, ad1, bg1, true);
    gat(bufB, wg2, as2, ad2, bg2, false);

    // pooling
    fill_f32<<<blocks((long)NG * H), 256, 0, stream>>>(g, 0.0f, (long)NG * H);
    fill_f32<<<blocks(NG), 256, 0, stream>>>(cnt, 0.0f, NG);
    pool_cnt<<<blocks(N), 256, 0, stream>>>(batch, cnt, N);
    pool_sum<<<blocks((long)N * H), 256, 0, stream>>>(bufB, batch, g, N, H);
    pool_div<<<blocks((long)NG * H), 256, 0, stream>>>(g, cnt, NG, H);

    // classifier
    small_gemm<<<blocks((long)NG * Hc), 256, 0, stream>>>(g, wc1, bc1, z, NG, H, Hc, 1);
    small_gemm<<<blocks((long)NG * NC), 256, 0, stream>>>(z, wc2, bc2, (float*)d_out, NG, Hc, NC, 0);
}
